// TtDeiTSelfAttention_10746008174892
// MI455X (gfx1250) — matrix operand order repaired and checked by reading, not compiled, not run on hardware
//
#include <hip/hip_runtime.h>
#include <hip/hip_bf16.h>

// DeiT self-attention on gfx1250:
//   Pass 0: fp32 -> f16 conversion of x and Wq/Wk/Wv (done once).
//   Pass 1: QKV projection with v_wmma_f32_16x16x32_f16, weight tile in LDS.
//   Pass 2: flash attention, 1 wave / 16-query tile, K/V tiles DMA'd to LDS
//           with TENSOR_LOAD_TO_LDS (double buffered, s_wait_tensorcnt).

typedef __attribute__((ext_vector_type(16))) _Float16 v16h;
typedef __attribute__((ext_vector_type(8)))  _Float16 v8h;
typedef __attribute__((ext_vector_type(8)))  float    v8f;
typedef __attribute__((ext_vector_type(4)))  float    v4f;
typedef __attribute__((ext_vector_type(4)))  unsigned int u32x4;
typedef __attribute__((ext_vector_type(8)))  int      i32x8;
typedef __attribute__((ext_vector_type(4)))  int      i32x4;

#define NB   8
#define SEQ  1024
#define HID  768
#define NH   12
#define HD   64
#define M_TOT (NB * SEQ)
#define QKV_ELEMS ((size_t)NB * NH * SEQ * HD)   // 6291456
#define X_ELEMS   ((size_t)M_TOT * HID)          // 6291456
#define W_ELEMS   ((size_t)HID * HID)            // 589824

__device__ __forceinline__ v8f wmma_f16(v16h a, v16h b, v8f c) {
  return __builtin_amdgcn_wmma_f32_16x16x32_f16(
      false, a, false, b, (short)0, c, false, false);
}

__device__ __forceinline__ float rmax16(float v) {
#pragma unroll
  for (int m = 1; m < 16; m <<= 1) v = fmaxf(v, __shfl_xor(v, m, 32));
  return v;
}
__device__ __forceinline__ float rsum16(float v) {
#pragma unroll
  for (int m = 1; m < 16; m <<= 1) v += __shfl_xor(v, m, 32);
  return v;
}

// --- TDM: 2D tile (16-bit elements) from global to LDS, D# per ISA 8.3/8.4 ---
__device__ __forceinline__ void tdm_load_tile_2d(
    unsigned lds_addr, const void* gptr,
    unsigned tensor_d0, unsigned tensor_d1, unsigned d0_stride,
    unsigned tile_d0, unsigned tile_d1,
    unsigned pad_interval_code, unsigned pad_amount_code)
{
  unsigned long long ga = (unsigned long long)(uintptr_t)gptr;
  u32x4 g0;
  g0[0] = 1u;                                    // count=1, user descriptor
  g0[1] = lds_addr;                              // LDS byte address
  g0[2] = (unsigned)(ga & 0xFFFFFFFFu);          // global_addr[31:0]
  g0[3] = (unsigned)((ga >> 32) & 0x1FFFFFFu)    // global_addr[56:32]
        | (2u << 30);                            // type = 2 (image)
  i32x8 g1;
  g1[0] = (int)((1u << 16)                       // data_size = 1 -> 2 bytes
        | (1u << 20)                             // pad_enable
        | (pad_interval_code << 22)
        | (pad_amount_code << 25));
  g1[1] = (int)((tensor_d0 & 0xFFFFu) << 16);            // dim0[15:0] @ 63:48
  g1[2] = (int)(((tensor_d0 >> 16) & 0xFFFFu)
        | ((tensor_d1 & 0xFFFFu) << 16));                // dim0[31:16], dim1[15:0]
  g1[3] = (int)(((tensor_d1 >> 16) & 0xFFFFu)
        | ((tile_d0 & 0xFFFFu) << 16));                  // dim1[31:16], tile0
  g1[4] = (int)(tile_d1 & 0xFFFFu);                      // tile1, tile2=0
  g1[5] = (int)d0_stride;                                // dim0 stride [31:0]
  g1[6] = 0;                                             // stride hi, dim1_stride lo
  g1[7] = 0;
  i32x4 gz = {};
#if defined(__clang_major__) && (__clang_major__ >= 23)
  i32x8 gz8 = {};
  __builtin_amdgcn_tensor_load_to_lds(g0, g1, gz, gz, gz8, 0);
#else
  __builtin_amdgcn_tensor_load_to_lds(g0, g1, gz, gz, 0);
#endif
}

// ---------------------------------------------------------------------------
// Pass 0: streaming fp32 -> f16 (8 elements / thread)
// ---------------------------------------------------------------------------
__global__ __launch_bounds__(256)
void cvt_f16_kernel(const float* __restrict__ src, _Float16* __restrict__ dst,
                    int n8)
{
  const int i = blockIdx.x * 256 + threadIdx.x;
  if (i >= n8) return;
  v4f a = *(const v4f*)(src + (size_t)i * 8);
  v4f b = *(const v4f*)(src + (size_t)i * 8 + 4);
  v8h h;
#pragma unroll
  for (int j = 0; j < 4; ++j) {
    h[j]     = (_Float16)a[j];
    h[4 + j] = (_Float16)b[j];
  }
  *(v8h*)(dst + (size_t)i * 8) = h;
}

// ---------------------------------------------------------------------------
// Pass 1: y = x @ W^T + b. Block = 8 waves, 128 rows x 16 channels.
// Weight tile (16 x 768 f16) staged in LDS, padded stride 776.
// ---------------------------------------------------------------------------
#define WLDS_STRIDE 776

__global__ __launch_bounds__(256)
void qkv_proj_kernel(const _Float16* __restrict__ xh,
                     const _Float16* __restrict__ wh,   // [3][768][768] f16
                     const float* __restrict__ bq, const float* __restrict__ bk,
                     const float* __restrict__ bv,
                     _Float16* __restrict__ q_ws, _Float16* __restrict__ k_ws,
                     _Float16* __restrict__ v_ws)
{
  __shared__ _Float16 wlds[16 * WLDS_STRIDE];

  const int which = blockIdx.z;
  const _Float16* W    = wh + (size_t)which * W_ELEMS;
  const float*    bias = (which == 0) ? bq : (which == 1) ? bk : bv;
  _Float16*       out  = (which == 0) ? q_ws : (which == 1) ? k_ws : v_ws;

  const int o0  = blockIdx.y * 16;
  const int tid = threadIdx.x;

  // cooperative stage of W[o0:o0+16, :] (f16, v8h granularity)
#pragma unroll 2
  for (int i8 = tid; i8 < (16 * HID) / 8; i8 += 256) {
    const int e   = i8 * 8;
    const int row = e / HID;
    const int col = e - row * HID;
    *(v8h*)&wlds[row * WLDS_STRIDE + col] =
        *(const v8h*)&W[(size_t)(o0 + row) * HID + col];
  }
  __syncthreads();

  const int wave  = tid >> 5;
  const int lane  = tid & 31;
  const int lhalf = lane >> 4;
  const int l16   = lane & 15;
  const int mrow  = blockIdx.x * 128 + wave * 16;
  const int arow  = mrow + l16;

  v8f c = {};
  for (int k0 = 0; k0 < HID; k0 += 32) {
    // A fragment: row arow, K = k0 + 8*lhalf + {0..7} and +16
    const _Float16* ap = xh + (size_t)arow * HID + k0 + lhalf * 8;
    v8h a0 = *(const v8h*)(ap);
    v8h a1 = *(const v8h*)(ap + 16);
    v16h a;
#pragma unroll
    for (int i = 0; i < 8; ++i) { a[i] = a0[i]; a[8 + i] = a1[i]; }
    // B fragment: col o0+l16, K = k0 + 8*lhalf + {0..7} and +16
    const _Float16* bp = &wlds[l16 * WLDS_STRIDE + k0 + lhalf * 8];
    v8h b0 = *(const v8h*)(bp);
    v8h b1 = *(const v8h*)(bp + 16);
    v16h b;
#pragma unroll
    for (int i = 0; i < 8; ++i) { b[i] = b0[i]; b[8 + i] = b1[i]; }
    c = wmma_f16(a, b, c);
  }

  const float bval = bias[o0 + l16];
  const int o = o0 + l16;
  const int h = o >> 6;
  const int d = o & 63;
#pragma unroll
  for (int r = 0; r < 8; ++r) {
    const int m = mrow + r + 8 * lhalf;
    const int b = m >> 10;
    const int s = m & (SEQ - 1);
    const _Float16 val = (_Float16)(c[r] + bval);
    if (which == 2)     // V transposed: [b, h, d, s]
      out[(((size_t)(b * NH + h)) * HD + d) * SEQ + s] = val;
    else                // Q, K: [b, h, s, d]
      out[(((size_t)(b * NH + h)) * SEQ + s) * HD + d] = val;
  }
}

// ---------------------------------------------------------------------------
// Pass 2: flash attention. 1 wave per block, 16 query rows x full 64-dim ctx.
// K/V tiles (32 keys) DMA'd into LDS by TDM, double buffered.
//   K tile: 32 rows x 64 f16, pad 4 dw / 32 dw  -> LDS stride 72 f16
//   V tile: 64 rows x 32 f16, pad 4 dw / 16 dw  -> LDS stride 40 f16
// ---------------------------------------------------------------------------
#define KB_STRIDE 72
#define VB_STRIDE 40
#define PLDS_STRIDE 40

__global__ __launch_bounds__(32)
void attn_kernel(const _Float16* __restrict__ q_ws,
                 const _Float16* __restrict__ k_ws,
                 const _Float16* __restrict__ v_ws,
                 float* __restrict__ out)
{
  __shared__ _Float16 kbuf[2][32 * KB_STRIDE];
  __shared__ _Float16 vbuf[2][64 * VB_STRIDE];
  __shared__ _Float16 plds[16 * PLDS_STRIDE];

  const int lane  = threadIdx.x & 31;
  const int lhalf = lane >> 4;
  const int l16   = lane & 15;
  const int q0    = blockIdx.x * 16;
  const int bh    = blockIdx.y;
  const int b     = bh / NH;
  const int h     = bh - b * NH;

  const _Float16* qbase = q_ws + (size_t)bh * SEQ * HD;
  const _Float16* kbase = k_ws + (size_t)bh * SEQ * HD;
  const _Float16* vbase = v_ws + (size_t)bh * HD * SEQ;   // [d][s]

  // Q fragments with 1/sqrt(64) folded in
  v16h aq0, aq1;
  {
    const _Float16* qp = qbase + (size_t)(q0 + l16) * HD + lhalf * 8;
    v8h x0 = *(const v8h*)(qp);
    v8h x1 = *(const v8h*)(qp + 16);
    v8h x2 = *(const v8h*)(qp + 32);
    v8h x3 = *(const v8h*)(qp + 48);
#pragma unroll
    for (int i = 0; i < 8; ++i) {
      aq0[i]     = (_Float16)((float)x0[i] * 0.125f);
      aq0[8 + i] = (_Float16)((float)x1[i] * 0.125f);
      aq1[i]     = (_Float16)((float)x2[i] * 0.125f);
      aq1[8 + i] = (_Float16)((float)x3[i] * 0.125f);
    }
  }

  float mrow[8], lrow[8];
  v8f acc[4] = {};
#pragma unroll
  for (int r = 0; r < 8; ++r) { mrow[r] = -1e30f; lrow[r] = 0.0f; }

  // prime first K/V tiles
  tdm_load_tile_2d((unsigned)(uintptr_t)&kbuf[0][0], kbase,
                   HD, SEQ, HD, HD, 32, /*interval 32dw*/4, /*4dw*/3);
  tdm_load_tile_2d((unsigned)(uintptr_t)&vbuf[0][0], vbase,
                   SEQ, HD, SEQ, 32, HD, /*interval 16dw*/3, /*4dw*/3);

  for (int kb = 0; kb < SEQ; kb += 32) {
    const int cur = (kb >> 5) & 1;
    const int nxt = cur ^ 1;
    if (kb + 32 < SEQ) {
      tdm_load_tile_2d((unsigned)(uintptr_t)&kbuf[nxt][0],
                       kbase + (size_t)(kb + 32) * HD,
                       HD, SEQ, HD, HD, 32, 4, 3);
      tdm_load_tile_2d((unsigned)(uintptr_t)&vbuf[nxt][0],
                       vbase + (kb + 32),
                       SEQ, HD, SEQ, 32, HD, 3, 3);
      __builtin_amdgcn_s_wait_tensorcnt(2);   // current pair complete
    } else {
      __builtin_amdgcn_s_wait_tensorcnt(0);
    }
    asm volatile("" ::: "memory");

    // ---- scores (two 16-key tiles) from LDS K buffer ----
    v8f s0 = {}, s1 = {};
#pragma unroll
    for (int nt = 0; nt < 2; ++nt) {
      v8f* st = nt ? &s1 : &s0;
#pragma unroll
      for (int kk = 0; kk < 2; ++kk) {
        const _Float16* kp =
            &kbuf[cur][(nt * 16 + l16) * KB_STRIDE + kk * 32 + lhalf * 8];
        v8h b0 = *(const v8h*)(kp);
        v8h b1 = *(const v8h*)(kp + 16);
        v16h bk;
#pragma unroll
        for (int i = 0; i < 8; ++i) { bk[i] = b0[i]; bk[8 + i] = b1[i]; }
        *st = wmma_f16(kk ? aq1 : aq0, bk, *st);
      }
    }

    // ---- online softmax over 32 new columns ----
    float corr[8];
#pragma unroll
    for (int r = 0; r < 8; ++r) {
      const float vmax = rmax16(fmaxf(s0[r], s1[r]));
      const float mn   = fmaxf(mrow[r], vmax);
      const float p0 = __builtin_exp2f((s0[r] - mn) * 1.442695041f);
      const float p1 = __builtin_exp2f((s1[r] - mn) * 1.442695041f);
      const float cf = __builtin_exp2f((mrow[r] - mn) * 1.442695041f);
      s0[r] = p0; s1[r] = p1;
      lrow[r] = lrow[r] * cf + rsum16(p0 + p1);
      mrow[r] = mn;
      corr[r] = cf;
    }
#pragma unroll
    for (int t = 0; t < 4; ++t)
#pragma unroll
      for (int r = 0; r < 8; ++r) acc[t][r] *= corr[r];

    // ---- P: C-layout -> LDS -> A-fragment ----
#pragma unroll
    for (int r = 0; r < 8; ++r) {
      const int row = r + 8 * lhalf;
      plds[row * PLDS_STRIDE + l16]      = (_Float16)s0[r];
      plds[row * PLDS_STRIDE + 16 + l16] = (_Float16)s1[r];
    }
    asm volatile("s_wait_dscnt 0" ::: "memory");
    v16h ap;
    {
      const _Float16* pp = &plds[l16 * PLDS_STRIDE + lhalf * 8];
      v8h p0v = *(const v8h*)(pp);
      v8h p1v = *(const v8h*)(pp + 16);
#pragma unroll
      for (int i = 0; i < 8; ++i) { ap[i] = p0v[i]; ap[8 + i] = p1v[i]; }
    }
    asm volatile("s_wait_dscnt 0" ::: "memory");

    // ---- ctx += P (16x32) @ V (32x64) from LDS V buffer ----
#pragma unroll
    for (int dt = 0; dt < 4; ++dt) {
      const _Float16* vp = &vbuf[cur][(dt * 16 + l16) * VB_STRIDE + lhalf * 8];
      v8h b0 = *(const v8h*)(vp);
      v8h b1 = *(const v8h*)(vp + 16);
      v16h bv;
#pragma unroll
      for (int i = 0; i < 8; ++i) { bv[i] = b0[i]; bv[8 + i] = b1[i]; }
      acc[dt] = wmma_f16(ap, bv, acc[dt]);
    }
  }

  // ---- epilogue ----
  float inv[8];
#pragma unroll
  for (int r = 0; r < 8; ++r) inv[r] = 1.0f / lrow[r];
#pragma unroll
  for (int dt = 0; dt < 4; ++dt) {
    const int d = dt * 16 + l16;
#pragma unroll
    for (int r = 0; r < 8; ++r) {
      const int srow = q0 + r + 8 * lhalf;
      out[((size_t)(b * SEQ + srow)) * HID + h * HD + d] = acc[dt][r] * inv[r];
    }
  }
}

// ---------------------------------------------------------------------------
extern "C" void kernel_launch(void* const* d_in, const int* in_sizes, int n_in,
                              void* d_out, int out_size, void* d_ws, size_t ws_size,
                              hipStream_t stream) {
  const float* hs = (const float*)d_in[0];
  const float* Wq = (const float*)d_in[1];
  const float* bq = (const float*)d_in[2];
  const float* Wk = (const float*)d_in[3];
  const float* bk = (const float*)d_in[4];
  const float* Wv = (const float*)d_in[5];
  const float* bv = (const float*)d_in[6];
  float* out = (float*)d_out;

  _Float16* qw = (_Float16*)d_ws;
  _Float16* kw = qw + QKV_ELEMS;
  _Float16* vw = kw + QKV_ELEMS;
  _Float16* xh = vw + QKV_ELEMS;
  _Float16* wh = xh + X_ELEMS;       // [3][768][768]

  // Pass 0: conversions (x, Wq, Wk, Wv)
  {
    const int n8x = (int)(X_ELEMS / 8);
    cvt_f16_kernel<<<dim3((n8x + 255) / 256), dim3(256), 0, stream>>>(hs, xh, n8x);
    const int n8w = (int)(W_ELEMS / 8);
    cvt_f16_kernel<<<dim3((n8w + 255) / 256), dim3(256), 0, stream>>>(Wq, wh, n8w);
    cvt_f16_kernel<<<dim3((n8w + 255) / 256), dim3(256), 0, stream>>>(Wk, wh + W_ELEMS, n8w);
    cvt_f16_kernel<<<dim3((n8w + 255) / 256), dim3(256), 0, stream>>>(Wv, wh + 2 * W_ELEMS, n8w);
  }

  // Pass 1: QKV projection
  dim3 g1(M_TOT / 128, HID / 16, 3), b1(256);
  qkv_proj_kernel<<<g1, b1, 0, stream>>>(xh, wh, bq, bk, bv, qw, kw, vw);

  // Pass 2: attention
  dim3 g2(SEQ / 16, NB * NH), b2(32);
  attn_kernel<<<g2, b2, 0, stream>>>(qw, kw, vw, out);
}
